// TopKRouter_69449621176583
// MI455X (gfx1250) — matrix operand order repaired and checked by reading, not compiled, not run on hardware
//
#include <hip/hip_runtime.h>
#include <hip/hip_bf16.h>

#define N_TOK 16384
#define C_DIM 4096
#define N_EXP 64
#define TOPK  2
#define CK    (C_DIM / 32)   // 128 k-chunks of 32 channels
#define ETILES 4             // 64 experts = 4 tiles of 16
#define KSPLIT 4             // C split into 4 segments -> 4096 GEMM waves
#define NCH   (CK / KSPLIT)  // 32 chunks per segment

typedef __attribute__((ext_vector_type(16))) __bf16 v16bf;
typedef __attribute__((ext_vector_type(8)))  float  v8f;

// fp32 -> bf16 bits, round-to-nearest-even
__device__ __forceinline__ unsigned short f32_to_bf16_bits(float f) {
    union { float f; unsigned u; } v; v.f = f;
    unsigned r = v.u + 0x7FFFu + ((v.u >> 16) & 1u);
    return (unsigned short)(r >> 16);
}

// ---------------------------------------------------------------------------
// Repack W (E x C fp32, row-major) into bf16 B-fragments.
// Wb layout: [(et*CK + ck)*32 + lane] * 16 elements, element i of lane l is
//   W[et*16 + (l&15)][ck*32 + (i&7) + ((i&8)?16:0) + ((l&16)?8:0)]
// (16-bit 32x16 B-matrix: lanes 0-15 hold K=0..7,16..23, lanes 16-31 hold K=8..15,24..31)
// ---------------------------------------------------------------------------
__global__ __launch_bounds__(256) void convert_W_kernel(
        const float* __restrict__ W, unsigned short* __restrict__ Wb) {
    int idx = blockIdx.x * 256 + threadIdx.x;          // 4*128*32*16 = 262144
    int i  = idx & 15;
    int l  = (idx >> 4) & 31;
    int ck = (idx >> 9) & 127;
    int et = idx >> 16;
    int e  = et * 16 + (l & 15);
    int k  = ck * 32 + (i & 7) + ((i & 8) ? 16 : 0) + ((l & 16) ? 8 : 0);
    Wb[idx] = f32_to_bf16_bits(W[(size_t)e * C_DIM + k]);
}

// ---------------------------------------------------------------------------
// partial[s][n][e] = sum over segment s of x[n][c]*W[e][c]  (bf16 WMMA, f32 acc)
// One wave: 16 tokens x 64 experts x 1024 channels. 4096 waves total.
// Software-pipelined: chunk i+1's A/B fragments are loaded while chunk i's
// WMMAs execute, so loadcnt waits don't serialize each v_wmma.
// ---------------------------------------------------------------------------
__global__ __launch_bounds__(256) void moe_logits_wmma(
        const float* __restrict__ x, const unsigned short* __restrict__ Wb,
        float* __restrict__ partial) {
    const int gw   = (int)((blockIdx.x * 256u + threadIdx.x) >> 5);  // 0..4095
    const int lane = threadIdx.x & 31;
    const int seg  = gw >> 10;          // K segment 0..3
    const int wv   = gw & 1023;         // token tile 0..1023
    const int row0 = wv << 4;
    const int m    = lane & 15;
    const int hi   = lane >> 4;
    const int ck0  = seg * NCH;
    const float* __restrict__ xrow =
        x + (size_t)(row0 + m) * C_DIM + ck0 * 32 + (hi ? 8 : 0);
    const v16bf* __restrict__ wb = (const v16bf*)Wb;

    v8f acc0 = (v8f){}, acc1 = (v8f){}, acc2 = (v8f){}, acc3 = (v8f){};

    // prologue: fetch chunk 0 fragments
    const float4* p0 = (const float4*)(xrow);
    float4 na0 = p0[0], na1 = p0[1], na2 = p0[4], na3 = p0[5];
    v16bf nB0 = wb[(size_t)(0 * CK + ck0) * 32 + lane];
    v16bf nB1 = wb[(size_t)(1 * CK + ck0) * 32 + lane];
    v16bf nB2 = wb[(size_t)(2 * CK + ck0) * 32 + lane];
    v16bf nB3 = wb[(size_t)(3 * CK + ck0) * 32 + lane];

    for (int i = 0; i < NCH; ++i) {
        float4 a0 = na0, a1 = na1, a2 = na2, a3 = na3;
        v16bf B0 = nB0, B1 = nB1, B2 = nB2, B3 = nB3;

        // issue next chunk's loads before this chunk's WMMAs
        int j = (i + 1 < NCH) ? i + 1 : i;     // last iter: harmless reload
        const float4* p = (const float4*)(xrow + j * 32);
        na0 = p[0]; na1 = p[1]; na2 = p[4]; na3 = p[5];
        nB0 = wb[(size_t)(0 * CK + ck0 + j) * 32 + lane];
        nB1 = wb[(size_t)(1 * CK + ck0 + j) * 32 + lane];
        nB2 = wb[(size_t)(2 * CK + ck0 + j) * 32 + lane];
        nB3 = wb[(size_t)(3 * CK + ck0 + j) * 32 + lane];
        int pf = (i + 4 < NCH) ? i + 4 : i;
        // WGP-scope regular-temporal prefetch: pull the line into near caches
        // (locality 3 -> SCOPE_WGP), consumed ~4 iterations later.
        __builtin_prefetch(xrow + pf * 32, 0, 3);   // global_prefetch_b8

        union { v16bf v; unsigned short s[16]; } A;
        A.s[0]  = f32_to_bf16_bits(a0.x); A.s[1]  = f32_to_bf16_bits(a0.y);
        A.s[2]  = f32_to_bf16_bits(a0.z); A.s[3]  = f32_to_bf16_bits(a0.w);
        A.s[4]  = f32_to_bf16_bits(a1.x); A.s[5]  = f32_to_bf16_bits(a1.y);
        A.s[6]  = f32_to_bf16_bits(a1.z); A.s[7]  = f32_to_bf16_bits(a1.w);
        A.s[8]  = f32_to_bf16_bits(a2.x); A.s[9]  = f32_to_bf16_bits(a2.y);
        A.s[10] = f32_to_bf16_bits(a2.z); A.s[11] = f32_to_bf16_bits(a2.w);
        A.s[12] = f32_to_bf16_bits(a3.x); A.s[13] = f32_to_bf16_bits(a3.y);
        A.s[14] = f32_to_bf16_bits(a3.z); A.s[15] = f32_to_bf16_bits(a3.w);

        acc0 = __builtin_amdgcn_wmma_f32_16x16x32_bf16(false, A.v, false, B0,
                                                       (short)0, acc0, false, false);
        acc1 = __builtin_amdgcn_wmma_f32_16x16x32_bf16(false, A.v, false, B1,
                                                       (short)0, acc1, false, false);
        acc2 = __builtin_amdgcn_wmma_f32_16x16x32_bf16(false, A.v, false, B2,
                                                       (short)0, acc2, false, false);
        acc3 = __builtin_amdgcn_wmma_f32_16x16x32_bf16(false, A.v, false, B3,
                                                       (short)0, acc3, false, false);
    }

    // C/D layout: acc[et][jj] holds M = jj + hi*8, N = lane&15
    float* __restrict__ out = partial + (size_t)seg * N_TOK * N_EXP;
    #pragma unroll
    for (int jj = 0; jj < 8; ++jj) {
        int M = jj + (hi << 3);
        size_t base = (size_t)(row0 + M) * N_EXP + m;
        out[base + 0]  = acc0[jj];
        out[base + 16] = acc1[jj];
        out[base + 32] = acc2[jj];
        out[base + 48] = acc3[jj];
    }
}

// ---------------------------------------------------------------------------
// Sum K-split partials (fixed order -> deterministic), softmax over 64,
// exact top-2 (tie -> lower index), per-wave register accumulation of
// importance / load counts, fixed-order block reduction. No atomics.
// ---------------------------------------------------------------------------
__global__ __launch_bounds__(256) void router_topk(
        const float* __restrict__ partial, float* __restrict__ out_scores,
        float* __restrict__ out_idx, float* __restrict__ blkImp,
        float* __restrict__ blkCnt) {
    __shared__ float shImp[8 * N_EXP];
    __shared__ float shCnt[8 * N_EXP];
    const int tid  = threadIdx.x;
    const int lane = tid & 31;
    const int wv   = tid >> 5;
    const int gwave = blockIdx.x * 8 + wv;   // 1024 waves total

    float impL = 0.0f, impH = 0.0f;   // importance partials: experts lane, lane+32
    float cntL = 0.0f, cntH = 0.0f;   // load-count partials

    for (int j = 0; j < 16; ++j) {
        const int t = gwave * 16 + j;
        float s0 = 0.0f, s1 = 0.0f;
        #pragma unroll
        for (int s = 0; s < KSPLIT; ++s) {   // fixed-order segment sum
            s0 += partial[((size_t)s * N_TOK + t) * N_EXP + lane];
            s1 += partial[((size_t)s * N_TOK + t) * N_EXP + 32 + lane];
        }
        // softmax
        float mx = fmaxf(s0, s1);
        #pragma unroll
        for (int o = 16; o > 0; o >>= 1) mx = fmaxf(mx, __shfl_xor(mx, o, 32));
        float e0 = __expf(s0 - mx), e1 = __expf(s1 - mx);
        float sum = e0 + e1;
        #pragma unroll
        for (int o = 16; o > 0; o >>= 1) sum += __shfl_xor(sum, o, 32);
        float inv = __frcp_rn(sum);
        float p0 = e0 * inv, p1 = e1 * inv;
        impL += p0;
        impH += p1;
        // top-1 (tie -> lower index)
        float v1 = p0; int i1 = lane;
        if (p1 > v1) { v1 = p1; i1 = lane + 32; }
        #pragma unroll
        for (int o = 16; o > 0; o >>= 1) {
            float ov = __shfl_xor(v1, o, 32); int oi = __shfl_xor(i1, o, 32);
            if (ov > v1 || (ov == v1 && oi < i1)) { v1 = ov; i1 = oi; }
        }
        // top-2: per-lane best excluding the global winner
        float v2; int i2;
        if (i1 == lane)           { v2 = p1; i2 = lane + 32; }
        else if (i1 == lane + 32) { v2 = p0; i2 = lane; }
        else if (p0 >= p1)        { v2 = p0; i2 = lane; }
        else                      { v2 = p1; i2 = lane + 32; }
        #pragma unroll
        for (int o = 16; o > 0; o >>= 1) {
            float ov = __shfl_xor(v2, o, 32); int oi = __shfl_xor(i2, o, 32);
            if (ov > v2 || (ov == v2 && oi < i2)) { v2 = ov; i2 = oi; }
        }
        // i1/i2 are wave-uniform: count selections into per-lane bins
        cntL += (i1 == lane) ? 1.0f : 0.0f;
        cntL += (i2 == lane) ? 1.0f : 0.0f;
        cntH += (i1 == lane + 32) ? 1.0f : 0.0f;
        cntH += (i2 == lane + 32) ? 1.0f : 0.0f;
        if (lane == 0) {
            out_scores[(size_t)t * TOPK + 0] = v1;
            out_scores[(size_t)t * TOPK + 1] = v2;
            out_idx[(size_t)t * TOPK + 0] = (float)i1;
            out_idx[(size_t)t * TOPK + 1] = (float)i2;
        }
    }

    // fixed-order per-block reduction (deterministic, no atomics)
    shImp[wv * N_EXP + lane]      = impL;
    shImp[wv * N_EXP + 32 + lane] = impH;
    shCnt[wv * N_EXP + lane]      = cntL;
    shCnt[wv * N_EXP + 32 + lane] = cntH;
    __syncthreads();
    if (tid < 64) {
        float s = 0.0f;
        #pragma unroll
        for (int w = 0; w < 8; ++w) s += shImp[w * N_EXP + tid];
        blkImp[(size_t)blockIdx.x * N_EXP + tid] = s;
    } else if (tid < 128) {
        int e = tid - 64;
        float s = 0.0f;
        #pragma unroll
        for (int w = 0; w < 8; ++w) s += shCnt[w * N_EXP + e];
        blkCnt[(size_t)blockIdx.x * N_EXP + e] = s;
    }
}

// aux = E * sum_e (S_e / N) * (C_e / (N*K)); fixed-order 128-block reduction
__global__ void finalize_aux(const float* __restrict__ blkImp,
                             const float* __restrict__ blkCnt,
                             float* __restrict__ out_aux) {
    int lane = threadIdx.x;   // 32 threads; lane handles experts lane, lane+32
    float iA = 0.0f, cA = 0.0f, iB = 0.0f, cB = 0.0f;
    for (int b = 0; b < 128; ++b) {
        iA += blkImp[b * N_EXP + lane];
        cA += blkCnt[b * N_EXP + lane];
        iB += blkImp[b * N_EXP + 32 + lane];
        cB += blkCnt[b * N_EXP + 32 + lane];
    }
    float part = iA * cA + iB * cB;
    #pragma unroll
    for (int o = 16; o > 0; o >>= 1) part += __shfl_xor(part, o, 32);
    if (lane == 0) {
        *out_aux = part * ((float)N_EXP) /
                   ((float)N_TOK * (float)N_TOK * (float)TOPK);
    }
}

extern "C" void kernel_launch(void* const* d_in, const int* in_sizes, int n_in,
                              void* d_out, int out_size, void* d_ws, size_t ws_size,
                              hipStream_t stream) {
    const float* x = (const float*)d_in[0];   // (N, C) fp32
    const float* W = (const float*)d_in[1];   // (E, C) fp32
    float* out = (float*)d_out;

    char* ws = (char*)d_ws;
    unsigned short* Wb = (unsigned short*)ws;                  // 512 KB bf16 B-frags
    float* partial = (float*)(ws + (size_t)512 * 1024);        // 4 x 4 MB = 16 MB
    float* blkImp  = (float*)(ws + (size_t)512 * 1024 + (size_t)16 * 1024 * 1024);
    float* blkCnt  = blkImp + 128 * N_EXP;                     // 32 KB each

    float* out_scores = out;                      // N*K
    float* out_idx    = out + N_TOK * TOPK;       // N*K (indices as float)
    float* out_aux    = out + 2 * N_TOK * TOPK;   // 1

    convert_W_kernel<<<(ETILES * CK * 32 * 16) / 256, 256, 0, stream>>>(W, Wb);
    moe_logits_wmma<<<(KSPLIT * N_TOK / 16) / 8, 256, 0, stream>>>(x, Wb, partial);
    router_topk<<<128, 256, 0, stream>>>(partial, out_scores, out_idx, blkImp, blkCnt);
    finalize_aux<<<1, 32, 0, stream>>>(blkImp, blkCnt, out_aux);
}
